// Attention_15925738734099
// MI455X (gfx1250) — matrix operand order repaired and checked by reading, not compiled, not run on hardware
//
#include <hip/hip_runtime.h>

typedef __attribute__((ext_vector_type(16))) __bf16 v16bf;
typedef __attribute__((ext_vector_type(8)))  float  v8f;

union ABf {
  v16bf v;
  unsigned int u[8];
  unsigned short s[16];
};

__device__ __forceinline__ unsigned short f2bf(float f) {
  union { float f; unsigned u; } c; c.f = f;
  unsigned r = c.u + 0x7FFFu + ((c.u >> 16) & 1u);   // round-to-nearest-even
  return (unsigned short)(r >> 16);
}
__device__ __forceinline__ float bf2f(unsigned short h) {
  union { float f; unsigned u; } c; c.u = ((unsigned)h) << 16;
  return c.f;
}

// A-operand (16x32 bf16, M=lane%16): two contiguous 8-element runs
//   u[0..3] = elements [kbase, kbase+7], u[4..7] = [kbase+16, kbase+23]
__device__ __forceinline__ void load_a16(ABf& a, const unsigned short* row, int kbase) {
  uint4 lo = *(const uint4*)(row + kbase);
  uint4 hi = *(const uint4*)(row + kbase + 16);
  a.u[0] = lo.x; a.u[1] = lo.y; a.u[2] = lo.z; a.u[3] = lo.w;
  a.u[4] = hi.x; a.u[5] = hi.y; a.u[6] = hi.z; a.u[7] = hi.w;
}
// B-operand (32x16 bf16, N-col = lane%16): k rows [kbase, kbase+7], [kbase+16, kbase+23]
__device__ __forceinline__ void load_b16(ABf& b, const unsigned short* col, int ld, int kbase) {
#pragma unroll
  for (int s = 0; s < 8; ++s) {
    b.s[s]     = col[(size_t)(kbase + s) * ld];
    b.s[8 + s] = col[(size_t)(kbase + 16 + s) * ld];
  }
}

#define XN   ((size_t)16384 * 512)   // x elements
#define WNE  ((size_t)512 * 512)     // one weight matrix

// ---------------------------------------------------------------------------
// Kernel 0: one-shot f32 -> bf16 conversion of x and Wq/Wk/Wv (bandwidth bound,
// ~35 MB in / ~17.5 MB out -> ~2.3 us at 23.3 TB/s).  8 elements per thread.
// ---------------------------------------------------------------------------
__global__ __launch_bounds__(256) void convert_kernel(
    const float* __restrict__ x,
    const float* __restrict__ Wq, const float* __restrict__ Wk,
    const float* __restrict__ Wv,
    unsigned short* __restrict__ xb,
    unsigned short* __restrict__ Wqb, unsigned short* __restrict__ Wkb,
    unsigned short* __restrict__ Wvb)
{
  size_t base = ((size_t)blockIdx.x * 256 + threadIdx.x) * 8;
  const float* src; unsigned short* dst;
  if (base < XN) { src = x + base; dst = xb + base; }
  else {
    size_t r = base - XN;
    int mat = (int)(r / WNE); size_t off = r % WNE;
    src = (mat == 0 ? Wq : mat == 1 ? Wk : Wv) + off;
    dst = (mat == 0 ? Wqb : mat == 1 ? Wkb : Wvb) + off;
  }
  float4 f0 = *(const float4*)(src);
  float4 f1 = *(const float4*)(src + 4);
  uint4 o;
  o.x = (unsigned)f2bf(f0.x) | ((unsigned)f2bf(f0.y) << 16);
  o.y = (unsigned)f2bf(f0.z) | ((unsigned)f2bf(f0.w) << 16);
  o.z = (unsigned)f2bf(f1.x) | ((unsigned)f2bf(f1.y) << 16);
  o.w = (unsigned)f2bf(f1.z) | ((unsigned)f2bf(f1.w) << 16);
  *(uint4*)dst = o;
}

// ---------------------------------------------------------------------------
// Kernel 1: Q/K/V projections from bf16 inputs.  b128-load + WMMA pipeline:
// the next k-step's A and all 4 B tiles are loaded while the current 4 WMMAs
// execute (fully unrolled -> rotation is pure SSA renaming, no v_mov).
// Q is written TRANSPOSED per batch (QbT[b][h][s]); K/V row-major.
// One wave = 16(M) x 64(N) strip; 1024*8*3 waves = 3072 blocks of 8 waves.
// ---------------------------------------------------------------------------
__global__ __launch_bounds__(256) void qkv_proj_kernel(
    const unsigned short* __restrict__ xb,
    const unsigned short* __restrict__ Wqb, const float* __restrict__ bq,
    const unsigned short* __restrict__ Wkb, const float* __restrict__ bk,
    const unsigned short* __restrict__ Wvb, const float* __restrict__ bv,
    unsigned short* __restrict__ QbT, unsigned short* __restrict__ Kb,
    unsigned short* __restrict__ Vb)
{
  const int lane = threadIdx.x & 31;
  const int wid  = blockIdx.x * 8 + (threadIdx.x >> 5);
  const int mat  = wid >> 13;           // / (1024*8)
  const int rem  = wid & 8191;
  const int m0   = (rem >> 3) << 4;     // 16-row tile over flattened [B*S]
  const int n0   = (rem & 7) << 6;      // 64-col strip over d_hid

  const unsigned short* W; const float* bias;
  if (mat == 0)      { W = Wqb; bias = bq; }
  else if (mat == 1) { W = Wkb; bias = bk; }
  else               { W = Wvb; bias = bv; }

  const int rowA  = lane & 15;
  const int kbase = (lane < 16) ? 0 : 8;
  const int colC  = lane & 15;
  const int rowC0 = (lane >> 4) << 3;

  v8f acc[4];
  for (int t = 0; t < 4; ++t)
    for (int i = 0; i < 8; ++i) acc[t][i] = 0.0f;

  const unsigned short* xr = xb + (size_t)(m0 + rowA) * 512;

  ABf a, b[4];
  load_a16(a, xr, kbase);
  {
    const unsigned short* wr = W + n0 + rowA;
#pragma unroll
    for (int t = 0; t < 4; ++t) load_b16(b[t], wr + 16 * t, 512, kbase);
  }

#pragma unroll
  for (int k0 = 0; k0 < 512; k0 += 32) {
    ABf an, bn[4];
    if (k0 + 32 < 512) {
      load_a16(an, xr + k0 + 32, kbase);
      const unsigned short* wr = W + (size_t)(k0 + 32) * 512 + n0 + rowA;
#pragma unroll
      for (int t = 0; t < 4; ++t) load_b16(bn[t], wr + 16 * t, 512, kbase);
      __builtin_prefetch(xr + k0 + 64, 0, 3);                        // WGP scope
      __builtin_prefetch(W + (size_t)(k0 + 64 + lane) * 512 + n0, 0, 3);
    } else {
      an = a;
#pragma unroll
      for (int t = 0; t < 4; ++t) bn[t] = b[t];
    }
#pragma unroll
    for (int t = 0; t < 4; ++t)
      acc[t] = __builtin_amdgcn_wmma_f32_16x16x32_bf16(
          false, a.v, false, b[t].v, (short)0, acc[t], false, false);
    a = an;
#pragma unroll
    for (int t = 0; t < 4; ++t) b[t] = bn[t];
  }

#pragma unroll
  for (int t = 0; t < 4; ++t) {
    const int n = n0 + 16 * t + colC;
    float bb = bias[n];
#pragma unroll
    for (int r = 0; r < 8; ++r) {
      const int m = m0 + rowC0 + r;
      unsigned short hv = f2bf(acc[t][r] + bb);
      if (mat == 0) {
        // transposed: QbT[b][h][s], per-batch stride 512*2048
        QbT[((size_t)(m >> 11) << 20) + (size_t)n * 2048 + (m & 2047)] = hv;
      } else if (mat == 1) {
        Kb[(size_t)m * 512 + n] = hv;
      } else {
        Vb[(size_t)m * 512 + n] = hv;
      }
    }
  }
}

// ---------------------------------------------------------------------------
// Kernel 2: flash-style attention.  One workgroup (8 waves) per
// (batch, 16-query tile).  Score strip 16x2048 bf16 in 64KB LDS.
// Phase 1 computes S^T = K x Q^T so the A operand (K rows) loads as 2x b128 (B is strided);
// all GEMM loops are software-pipelined by one k-step.
// K/V/Q^T (48MB bf16) all fit in the 192MB L2 -> re-reads never hit HBM.
// ---------------------------------------------------------------------------
__global__ __launch_bounds__(256) void flash_attn_kernel(
    const unsigned short* __restrict__ QbT,
    const unsigned short* __restrict__ Kb,
    const unsigned short* __restrict__ Vb,
    float* __restrict__ out)
{
  __shared__ unsigned short s_scores[16 * 2048];   // 64 KB

  const int lane = threadIdx.x & 31;
  const int wave = threadIdx.x >> 5;
  const int b    = blockIdx.x >> 7;     // batch
  const int q0   = (blockIdx.x & 127) * 16;

  const unsigned short* QT = QbT + ((size_t)b << 20);         // [512][2048]
  const unsigned short* K  = Kb  + (size_t)b * 2048 * 512;
  const unsigned short* V  = Vb  + (size_t)b * 2048 * 512;
  float* O = out + (size_t)b * 2048 * 512;

  const int rowA  = lane & 15;
  const int kbase = (lane < 16) ? 0 : 8;
  const int colC  = lane & 15;
  const int rowC0 = (lane >> 4) << 3;
  const float scale = 0.044194173824159216f;   // 1/sqrt(512)

  // ---- Phase 1: S^T tile = K_tile(16k x 32h) x Q^T(32h x 16q).
  // This wave covers keys [wave*256, +256).
  for (int kt = 0; kt < 16; ++kt) {
    const int key0 = wave * 256 + kt * 16;
    v8f acc; for (int i = 0; i < 8; ++i) acc[i] = 0.0f;
    const unsigned short* kr = K + (size_t)(key0 + rowA) * 512;

    ABf a, bm;
    load_a16(a, kr, kbase);                           // A = K rows
    load_b16(bm, QT + q0 + rowA, 2048, kbase);      // B = Q^T rows
#pragma unroll
    for (int h0 = 0; h0 < 512; h0 += 32) {
      ABf an, bn;
      if (h0 + 32 < 512) {
        load_a16(an, kr + h0 + 32, kbase);
        load_b16(bn, QT + (size_t)(h0 + 32) * 2048 + q0 + rowA, 2048, kbase);
      } else { an = a; bn = bm; }
      acc = __builtin_amdgcn_wmma_f32_16x16x32_bf16(
          false, a.v, false, bm.v, (short)0, acc, false, false);
      a = an; bm = bn;
    }
    // warm L0 with the next key tile's K rows (WGP scope)
    if (kt + 1 < 16)
      __builtin_prefetch(K + (size_t)(key0 + 16 + rowA) * 512, 0, 3);
    // D = S^T: row = key, col = query -> scores[query][key]
#pragma unroll
    for (int r = 0; r < 8; ++r)
      s_scores[colC * 2048 + key0 + rowC0 + r] = f2bf(acc[r] * scale);
  }
  __syncthreads();

  // ---- Phase 2: row softmax.  16 threads per row; shfl_xor butterflies
  // stay inside each 16-lane half of a wave32.
  {
    const int r = threadIdx.x >> 4;
    const int j = threadIdx.x & 15;
    unsigned short* srow = s_scores + r * 2048;
    float mx = -3.4e38f;
    for (int i = 0; i < 128; ++i)
      mx = fmaxf(mx, bf2f(srow[j + 16 * i]));
    mx = fmaxf(mx, __shfl_xor(mx, 1));
    mx = fmaxf(mx, __shfl_xor(mx, 2));
    mx = fmaxf(mx, __shfl_xor(mx, 4));
    mx = fmaxf(mx, __shfl_xor(mx, 8));
    float sum = 0.0f;
    for (int i = 0; i < 128; ++i)
      sum += __expf(bf2f(srow[j + 16 * i]) - mx);
    sum += __shfl_xor(sum, 1);
    sum += __shfl_xor(sum, 2);
    sum += __shfl_xor(sum, 4);
    sum += __shfl_xor(sum, 8);
    const float inv = 1.0f / sum;
    for (int i = 0; i < 128; ++i) {
      int c = j + 16 * i;
      srow[c] = f2bf(__expf(bf2f(srow[c]) - mx) * inv);
    }
  }
  __syncthreads();

  // ---- Phase 3: O = P(16q x 2048k) x V(2048k x 512h);
  // this wave covers h-cols [wave*64, +64).  Pipelined by one k-step;
  // unroll x2 so the operand rotation is SSA renaming, not v_movs.
  const int n0 = wave * 64;
  v8f acc4[4];
  for (int t = 0; t < 4; ++t)
    for (int i = 0; i < 8; ++i) acc4[t][i] = 0.0f;

  ABf pa, pb[4];
  load_a16(pa, s_scores + rowA * 2048, kbase);        // A = P (2x ds_load_b128)
  {
    const unsigned short* vr = V + n0 + rowA;
#pragma unroll
    for (int t = 0; t < 4; ++t) load_b16(pb[t], vr + 16 * t, 512, kbase);
  }

#pragma unroll 2
  for (int ks = 0; ks < 64; ++ks) {
    const int key0 = ks * 32;
    ABf an, bn[4];
    if (ks + 1 < 64) {
      load_a16(an, s_scores + rowA * 2048 + key0 + 32, kbase);
      const unsigned short* vr = V + (size_t)(key0 + 32) * 512 + n0 + rowA;
#pragma unroll
      for (int t = 0; t < 4; ++t) load_b16(bn[t], vr + 16 * t, 512, kbase);
      __builtin_prefetch(V + (size_t)(key0 + 64 + lane) * 512 + n0, 0, 3);
    } else {
      an = pa;
#pragma unroll
      for (int t = 0; t < 4; ++t) bn[t] = pb[t];
    }
#pragma unroll
    for (int t = 0; t < 4; ++t)
      acc4[t] = __builtin_amdgcn_wmma_f32_16x16x32_bf16(
          false, pa.v, false, pb[t].v, (short)0, acc4[t], false, false);
    pa = an;
#pragma unroll
    for (int t = 0; t < 4; ++t) pb[t] = bn[t];
  }

#pragma unroll
  for (int t = 0; t < 4; ++t)
#pragma unroll
    for (int r = 0; r < 8; ++r)
      O[(size_t)(q0 + rowC0 + r) * 512 + n0 + 16 * t + colC] = acc4[t][r];
}

// ---------------------------------------------------------------------------
extern "C" void kernel_launch(void* const* d_in, const int* in_sizes, int n_in,
                              void* d_out, int out_size, void* d_ws, size_t ws_size,
                              hipStream_t stream) {
  const float* x  = (const float*)d_in[0];
  const float* Wq = (const float*)d_in[1];
  const float* bq = (const float*)d_in[2];
  const float* Wk = (const float*)d_in[3];
  const float* bk = (const float*)d_in[4];
  const float* Wv = (const float*)d_in[5];
  const float* bv = (const float*)d_in[6];
  float* out = (float*)d_out;

  // workspace layout (bf16 = unsigned short)
  unsigned short* xb  = (unsigned short*)d_ws;   // 16 MB
  unsigned short* QbT = xb  + XN;                // 16 MB  (transposed Q)
  unsigned short* Kb  = QbT + XN;                // 16 MB
  unsigned short* Vb  = Kb  + XN;                // 16 MB
  unsigned short* Wqb = Vb  + XN;                // 0.5 MB
  unsigned short* Wkb = Wqb + WNE;               // 0.5 MB
  unsigned short* Wvb = Wkb + WNE;               // 0.5 MB

  // (XN + 3*WNE) / 8 elements-per-thread / 256 threads-per-block = 4480 blocks
  convert_kernel<<<4480, 256, 0, stream>>>(x, Wq, Wk, Wv, xb, Wqb, Wkb, Wvb);
  // 1024 Mtiles * 8 Nstrips * 3 matrices = 24576 waves / 8 per block
  qkv_proj_kernel<<<3072, 256, 0, stream>>>(xb, Wqb, bq, Wkb, bk, Wvb, bv,
                                            QbT, Kb, Vb);
  // 8 batches * 128 query tiles
  flash_attn_kernel<<<1024, 256, 0, stream>>>(QbT, Kb, Vb, out);
}
